// SDFRenderer_27066883900026
// MI455X (gfx1250) — matrix operand order repaired and checked by reading, not compiled, not run on hardware
//
#include <hip/hip_runtime.h>
#include <hip/hip_bf16.h>
#include <math.h>

// ---------------- CDNA5 WMMA types ----------------
typedef __attribute__((ext_vector_type(16))) _Float16   v16h;
typedef __attribute__((ext_vector_type(2)))  __fp16     v2g;   // cvt_pkrtz result type
typedef __attribute__((ext_vector_type(8)))  float      v8f;
typedef __attribute__((ext_vector_type(4)))  unsigned   v4u;
typedef __attribute__((ext_vector_type(2)))  unsigned   v2u;
typedef __attribute__((ext_vector_type(8)))  int        v8i;
typedef __attribute__((ext_vector_type(4)))  int        v4i;

#define IMAGE_SIZE 512
#define N_RAYS     (IMAGE_SIZE * IMAGE_SIZE)
#define HIDDEN     32
#define ITERS      20
#define RENDER_STEP 0.95f
#define HIT_THRESH  0.05f
#define NEARF       0.1f
#define FARF        255.0f

// LDS activation tile: 16 rays x 32 f16, padded to 40 halves (80B) per row
#define ROWH 40
#define TILEH (16 * ROWH)

// LDS weight staging offsets (floats)
#define OFF_W1  0      // 96
#define OFF_B1  96     // 32
#define OFF_W2  128    // 1024
#define OFF_B2  1152   // 32
#define OFF_W3  1184   // 32
#define OFF_B3  1216   // 1
#define OFF_TW1 1220   // 96
#define OFF_TB1 1316   // 32
#define OFF_TW2 1348   // 96
#define OFF_TB2 1444   // 3
#define WLDS_SZ 1448

union HV16 { v4u q[2]; _Float16 h[16]; v16h v; };
union PK   { v2g v; _Float16 h[2]; unsigned u; };   // reinterpret __fp16x2 lanes

__device__ __forceinline__ void lds_fence() {
  // DS ops within a wave are in-order; this makes cross-lane LDS stores
  // visible before reads and stops compiler reordering.
  asm volatile("s_wait_dscnt 0" ::: "memory");
}

__device__ __forceinline__ v8f wmma_f16(v16h a, v16h b, v8f c) {
  return __builtin_amdgcn_wmma_f32_16x16x32_f16(
      false, a, false, b, (short)0, c, false, false);
}

// 1-op ReLU: v_med3_f32(x, 0, +inf)
__device__ __forceinline__ float relu(float x) {
  return __builtin_amdgcn_fmed3f(x, 0.0f, __builtin_inff());
}

// tanh via exp path: 1 trans + few VALU, NaN/overflow safe (arg <= 0)
__device__ __forceinline__ float fast_tanh(float x) {
  const float e = __expf(-2.0f * fabsf(x));
  const float r = (1.0f - e) * __builtin_amdgcn_rcpf(1.0f + e);
  return copysignf(r, x);
}

__device__ __forceinline__ float fast_sigmoid(float x) {
  return __builtin_amdgcn_rcpf(1.0f + __expf(-x));
}

// A-matrix (16x32 f16): lane L owns row M=L&15.
// lanes 0-15 : halves = K 0..7  and K 16..23
// lanes 16-31: halves = K 8..15 and K 24..31
__device__ __forceinline__ v16h load_A(const _Float16* row, int koffA /*0|8*/) {
  HV16 u;
  u.q[0] = *(const v4u*)(row + koffA);
  u.q[1] = *(const v4u*)(row + 16 + koffA);
  return u.v;
}

// A-half index t -> K index for this lane
__device__ __forceinline__ int kmap(int t, int koffA) {
  return (t < 8) ? (koffA + t) : (16 + koffA + (t - 8));
}

// B-matrix (32x16 f16): lane L owns column N; halves t=0..15 are
// K = kmap(t, lane>=16 ? 8 : 0).  W is Krows x 32 (f32, LDS).
__device__ __forceinline__ v16h build_B(const float* W, int Krows, int N, int koffB) {
  HV16 u;
#pragma unroll
  for (int t = 0; t < 16; ++t) {
    int K = kmap(t, koffB);
    float w = (K < Krows) ? W[K * HIDDEN + N] : 0.0f;
    u.h[t] = (_Float16)w;
  }
  return u.v;
}

__device__ __forceinline__ v8f bias_C(float b) {
  v8f c;
#pragma unroll
  for (int r = 0; r < 8; ++r) c[r] = b;
  return c;
}

// Write x,y,z into K=0..3 of the activation row (one b64). K>=3 columns of
// B1/tW1 are exact zeros, so stale halves at K>=4 contribute 0 to the WMMA.
__device__ __forceinline__ void store_x_row(_Float16* row, float x0, float x1, float x2) {
  PK a, b;
  a.v = __builtin_amdgcn_cvt_pkrtz(x0, x1);
  b.v = __builtin_amdgcn_cvt_pkrtz(x2, 0.0f);
  union { v2u q; unsigned u[2]; } w;
  w.u[0] = a.u; w.u[1] = b.u;
  *(v2u*)row = w.q;
}

// Scatter D (two 16x16 f32 tiles) with ReLU into the f16 activation tile.
// v_med3 relu + v_cvt_pk_rtz pack, then b16 / b16_d16_hi store pair.
__device__ __forceinline__ void store_h(_Float16* tile, v8f c0, v8f c1, int mh, int n0) {
#pragma unroll
  for (int r = 0; r < 8; ++r) {
    int M = r + mh;
    PK pk;
    pk.v = __builtin_amdgcn_cvt_pkrtz(relu(c0[r]), relu(c1[r]));
    tile[M * ROWH + n0]      = pk.h[0];
    tile[M * ROWH + n0 + 16] = pk.h[1];
  }
}

// 16-half partial dot of an A-chunk against per-lane weight registers
__device__ __forceinline__ float dot16(v16h a, const float* wloc) {
  float acc = 0.0f;
#pragma unroll
  for (int t = 0; t < 16; ++t) acc += (float)a[t] * wloc[t];
  return acc;
}

// -------------------------------------------------------------------------
// Pass 1: sphere trace + texture MLP.
// 256 thr = 8 waves; each wave traces TWO 16-ray tiles (32 rays) so each
// layer issues 4 independent WMMAs (fills the WMMA->VALU hazard slots).
// dW2 (32x32 f32) is staged into LDS by the Tensor Data Mover.
// -------------------------------------------------------------------------
__global__ __launch_bounds__(256) void sdf_pass1(
    const float* __restrict__ eye, const float* __restrict__ cam,
    const float* __restrict__ rays,
    const float* __restrict__ dW1, const float* __restrict__ db1,
    const float* __restrict__ dW2, const float* __restrict__ db2,
    const float* __restrict__ dW3, const float* __restrict__ db3,
    const float* __restrict__ tW1, const float* __restrict__ tb1,
    const float* __restrict__ tW2, const float* __restrict__ tb2,
    float* __restrict__ ws, unsigned* __restrict__ scal) {
  __shared__ __align__(16) float    wsm[WLDS_SZ];
  __shared__ __align__(16) _Float16 act[8 * 2 * TILEH];

  // ---- dW2 via TDM: 2D tensor 32x32 f32, tile 32x32, stride 32 ----
#if __has_builtin(__builtin_amdgcn_tensor_load_to_lds)
  if (threadIdx.x < 32) {   // one wave issues the DMA (EXEC ignored by TDM)
    const unsigned long long ga = (unsigned long long)(uintptr_t)dW2;
    const unsigned lds_addr = (unsigned)(uintptr_t)(&wsm[OFF_W2]); // low bits = LDS offset
    v4u g0; v8i g1; v4i g2; v4i g3;
    g0[0] = 1u;                                   // count=1, no gather
    g0[1] = lds_addr;                             // lds_addr[31:0]
    g0[2] = (unsigned)(ga & 0xFFFFFFFFu);         // global_addr[31:0]
    g0[3] = (unsigned)((ga >> 32) & 0x01FFFFFFu)  // global_addr[56:32]
          | (2u << 30);                           // type=2 (image)
    g1[0] = (2 << 16);                            // data_size=2 (4B), mask=0
    g1[1] = (32 << 16);                           // tensor_dim0[15:0]=32
    g1[2] = (32 << 16);                           // dim0 hi=0 | tensor_dim1 lo=32
    g1[3] = (32 << 16);                           // dim1 hi=0 | tile_dim0=32
    g1[4] = 32;                                   // tile_dim1=32, tile_dim2=0
    g1[5] = 32;                                   // tensor_dim0_stride[31:0]=32
    g1[6] = 0; g1[7] = 0;                         // stride hi, dim1_stride=0 (2D)
    g2[0] = g2[1] = g2[2] = g2[3] = 0;
    g3[0] = g3[1] = g3[2] = g3[3] = 0;
#if __clang_major__ >= 23
    v8i g4; g4[0]=g4[1]=g4[2]=g4[3]=g4[4]=g4[5]=g4[6]=g4[7]=0;
    __builtin_amdgcn_tensor_load_to_lds(g0, g1, g2, g3, g4, 0);
#else
    __builtin_amdgcn_tensor_load_to_lds(g0, g1, g2, g3, 0);
#endif
#if __has_builtin(__builtin_amdgcn_s_wait_tensorcnt)
    __builtin_amdgcn_s_wait_tensorcnt(0);
#else
    asm volatile("s_wait_tensorcnt 0" ::: "memory");
#endif
  }
#else
  for (int t = threadIdx.x; t < 1024; t += 256) wsm[OFF_W2 + t] = dW2[t];
#endif

  // ---- stage the small weights cooperatively ----
  for (int t = threadIdx.x; t < 96;   t += 256) wsm[OFF_W1  + t] = dW1[t];
  for (int t = threadIdx.x; t < 32;   t += 256) wsm[OFF_B1  + t] = db1[t];
  for (int t = threadIdx.x; t < 32;   t += 256) wsm[OFF_B2  + t] = db2[t];
  for (int t = threadIdx.x; t < 32;   t += 256) wsm[OFF_W3  + t] = dW3[t];
  for (int t = threadIdx.x; t < 1;    t += 256) wsm[OFF_B3  + t] = db3[t];
  for (int t = threadIdx.x; t < 96;   t += 256) wsm[OFF_TW1 + t] = tW1[t];
  for (int t = threadIdx.x; t < 32;   t += 256) wsm[OFF_TB1 + t] = tb1[t];
  for (int t = threadIdx.x; t < 96;   t += 256) wsm[OFF_TW2 + t] = tW2[t];
  for (int t = threadIdx.x; t < 3;    t += 256) wsm[OFF_TB2 + t] = tb2[t];
  __syncthreads();

  const int lane  = threadIdx.x & 31;
  const int wave  = threadIdx.x >> 5;
  const int n0    = lane & 15;
  const int koffA = (lane & 16) ? 8  : 0;
  const int koffB = (lane & 16) ? 8  : 0;
  const int mh    = (lane & 16) ? 8  : 0;
  _Float16* tileA = act + wave * (2 * TILEH);
  _Float16* tileB = tileA + TILEH;
  _Float16* rowA  = tileA + n0 * ROWH;
  _Float16* rowB  = tileB + n0 * ROWH;

  // B-format weight registers, resident for all 20 iterations
  const v16h B1lo = build_B(&wsm[OFF_W1], 3,      n0,      koffB);
  const v16h B1hi = build_B(&wsm[OFF_W1], 3,      n0 + 16, koffB);
  const v16h B2lo = build_B(&wsm[OFF_W2], HIDDEN, n0,      koffB);
  const v16h B2hi = build_B(&wsm[OFF_W2], HIDDEN, n0 + 16, koffB);
  const float b1lo = wsm[OFF_B1 + n0], b1hi = wsm[OFF_B1 + n0 + 16];
  const float b2lo = wsm[OFF_B2 + n0], b2hi = wsm[OFF_B2 + n0 + 16];
  const float b3   = wsm[OFF_B3];

  // W3 gathered in A-half order for the in-register layer-3 dot
  float w3loc[16];
#pragma unroll
  for (int t = 0; t < 16; ++t) w3loc[t] = wsm[OFF_W3 + kmap(t, koffA)];

  // Two rays per lane-slot (lanes 16-31 mirror 0-15 -> no divergent math)
  const int ri0 = blockIdx.x * 256 + wave * 32 + n0;
  const int ri1 = ri0 + 16;
  const float e0 = eye[0], e1 = eye[1], e2 = eye[2];

  float rtA0, rtA1, rtA2, rtB0, rtB1, rtB2;
  {
    float r0 = rays[ri0 * 3 + 0], r1 = rays[ri0 * 3 + 1], r2 = rays[ri0 * 3 + 2];
    rtA0 = r0 * cam[0] + r1 * cam[3] + r2 * cam[6];
    rtA1 = r0 * cam[1] + r1 * cam[4] + r2 * cam[7];
    rtA2 = r0 * cam[2] + r1 * cam[5] + r2 * cam[8];
    r0 = rays[ri1 * 3 + 0]; r1 = rays[ri1 * 3 + 1]; r2 = rays[ri1 * 3 + 2];
    rtB0 = r0 * cam[0] + r1 * cam[3] + r2 * cam[6];
    rtB1 = r0 * cam[1] + r1 * cam[4] + r2 * cam[7];
    rtB2 = r0 * cam[2] + r1 * cam[5] + r2 * cam[8];
  }

  float xA0 = e0, xA1 = e1, xA2 = e2, xB0 = e0, xB1 = e1, xB2 = e2;
  float pA0 = e0, pA1 = e1, pA2 = e2, pB0 = e0, pB1 = e1, pB2 = e2;

#pragma unroll 1
  for (int it = 0; it < ITERS; ++it) {
    // ---- layer 1: x (K=0..2, zero-padded cols in B1) -> h1(32) ----
    if (lane < 16) {
      store_x_row(rowA, xA0, xA1, xA2);
      store_x_row(rowB, xB0, xB1, xB2);
    }
    lds_fence();
    v16h aA = load_A(rowA, koffA);
    v16h aB = load_A(rowB, koffA);
    v8f cA0 = wmma_f16(aA, B1lo, bias_C(b1lo));
    v8f cA1 = wmma_f16(aA, B1hi, bias_C(b1hi));
    v8f cB0 = wmma_f16(aB, B1lo, bias_C(b1lo));
    v8f cB1 = wmma_f16(aB, B1hi, bias_C(b1hi));
    store_h(tileA, cA0, cA1, mh, n0);
    store_h(tileB, cB0, cB1, mh, n0);
    lds_fence();

    // ---- layer 2: h1 @ W2 -> h2(32) ----
    aA  = load_A(rowA, koffA);
    aB  = load_A(rowB, koffA);
    cA0 = wmma_f16(aA, B2lo, bias_C(b2lo));
    cA1 = wmma_f16(aA, B2hi, bias_C(b2hi));
    cB0 = wmma_f16(aB, B2lo, bias_C(b2lo));
    cB1 = wmma_f16(aB, B2hi, bias_C(b2hi));
    store_h(tileA, cA0, cA1, mh, n0);
    store_h(tileB, cB0, cB1, mh, n0);
    lds_fence();

    // ---- layer 3: tanh(h2 . W3 + b3).  Each lane holds 16 of the 32 row
    // halves in A-format; partner lane (lane^16) holds the rest. ----
    aA = load_A(rowA, koffA);
    aB = load_A(rowB, koffA);
    float pa = dot16(aA, w3loc);
    float pb = dot16(aB, w3loc);
    pa += __shfl_xor(pa, 16);
    pb += __shfl_xor(pb, 16);
    const float dA = fast_tanh(pa + b3);
    const float dB = fast_tanh(pb + b3);

    if (it == ITERS - 1) {
      pA0 = xA0; pA1 = xA1; pA2 = xA2;
      pB0 = xB0; pB1 = xB1; pB2 = xB2;
    }
    const float tA = dA * RENDER_STEP, tB = dB * RENDER_STEP;
    xA0 += rtA0 * tA; xA1 += rtA1 * tA; xA2 += rtA2 * tA;
    xB0 += rtB0 * tB; xB1 += rtB1 * tB; xB2 += rtB2 * tB;
  }

  // ---- hit / distance bookkeeping ----
  float d0 = xA0 - pA0, d1 = xA1 - pA1, d2 = xA2 - pA2;
  const float distA = sqrtf(d0 * d0 + d1 * d1 + d2 * d2);
  d0 = xB0 - pB0; d1 = xB1 - pB1; d2 = xB2 - pB2;
  const float distB = sqrtf(d0 * d0 + d1 * d1 + d2 * d2);
  d0 = xA0 - e0; d1 = xA1 - e1; d2 = xA2 - e2;
  const float lenA = sqrtf(d0 * d0 + d1 * d1 + d2 * d2);
  d0 = xB0 - e0; d1 = xB1 - e1; d2 = xB2 - e2;
  const float lenB = sqrtf(d0 * d0 + d1 * d1 + d2 * d2);
  const bool hitA = fabsf(distA) < HIT_THRESH;
  const bool hitB = fabsf(distB) < HIT_THRESH;
  const float sA = fminf(fmaxf(lenA, NEARF), FARF);
  const float sB = fminf(fmaxf(lenB, NEARF), FARF);

  // ---- texture MLP: th = relu(x@tW1+tb1); rgb = sigmoid(th@tW2+tb2) ----
  if (lane < 16) {
    store_x_row(rowA, xA0, xA1, xA2);
    store_x_row(rowB, xB0, xB1, xB2);
  }
  lds_fence();
  {
    const v16h Tlo = build_B(&wsm[OFF_TW1], 3, n0,      koffB);
    const v16h Thi = build_B(&wsm[OFF_TW1], 3, n0 + 16, koffB);
    const float tb1lo = wsm[OFF_TB1 + n0], tb1hi = wsm[OFF_TB1 + n0 + 16];
    v16h aA = load_A(rowA, koffA);
    v16h aB = load_A(rowB, koffA);
    v8f cA0 = wmma_f16(aA, Tlo, bias_C(tb1lo));
    v8f cA1 = wmma_f16(aA, Thi, bias_C(tb1hi));
    v8f cB0 = wmma_f16(aB, Tlo, bias_C(tb1lo));
    v8f cB1 = wmma_f16(aB, Thi, bias_C(tb1hi));
    store_h(tileA, cA0, cA1, mh, n0);
    store_h(tileB, cB0, cB1, mh, n0);
    lds_fence();
  }
  float rgbA[3], rgbB[3];
  {
    const v16h aA = load_A(rowA, koffA);
    const v16h aB = load_A(rowB, koffA);
#pragma unroll
    for (int c = 0; c < 3; ++c) {
      float pa = 0.0f, pb = 0.0f;
#pragma unroll
      for (int t = 0; t < 16; ++t) {
        const float w = wsm[OFF_TW2 + kmap(t, koffA) * 3 + c];
        pa += (float)aA[t] * w;
        pb += (float)aB[t] * w;
      }
      pa += __shfl_xor(pa, 16);
      pb += __shfl_xor(pb, 16);
      rgbA[c] = fast_sigmoid(pa + wsm[OFF_TB2 + c]);
      rgbB[c] = fast_sigmoid(pb + wsm[OFF_TB2 + c]);
    }
  }

  // ---- per-ray outputs ----
  if (lane < 16) {
    ws[ri0 * 4 + 0] = rgbA[0]; ws[ri0 * 4 + 1] = rgbA[1];
    ws[ri0 * 4 + 2] = rgbA[2]; ws[ri0 * 4 + 3] = hitA ? sA : -1.0f;
    ws[ri1 * 4 + 0] = rgbB[0]; ws[ri1 * 4 + 1] = rgbB[1];
    ws[ri1 * 4 + 2] = rgbB[2]; ws[ri1 * 4 + 3] = hitB ? sB : -1.0f;
  }

  // ---- wave reduction -> uint-encoded float atomics ----
  float smax_c = fmaxf(hitA ? sA : 0.0f,    hitB ? sB : 0.0f);
  float smin_c = fminf(hitA ? sA : 3.0e38f, hitB ? sB : 3.0e38f);
  unsigned hb  = (hitA || hitB) ? 1u : 0u;
#pragma unroll
  for (int m = 16; m >= 1; m >>= 1) {
    smax_c = fmaxf(smax_c, __shfl_xor(smax_c, m));
    smin_c = fminf(smin_c, __shfl_xor(smin_c, m));
    hb    |= (unsigned)__shfl_xor((int)hb, m);
  }
  if (lane == 0) {
    atomicMax(&scal[0], __float_as_uint(smax_c));
    atomicMin(&scal[1], __float_as_uint(smin_c));
    if (hb) atomicOr(&scal[2], 1u);
  }
}

// -------------------------------------------------------------------------
// Pass 2: apply global 'distance' shading, write planar [3,H,W] output.
// -------------------------------------------------------------------------
__global__ __launch_bounds__(256) void sdf_pass2(
    const float* __restrict__ ws, const unsigned* __restrict__ scal,
    const float* __restrict__ bg, float* __restrict__ out) {
  const int i = blockIdx.x * 256 + threadIdx.x;
  if (i >= N_RAYS) return;
  const float s4   = ws[i * 4 + 3];
  const bool  hit  = s4 > 0.0f;
  const bool  anyh = scal[2] != 0u;
  const float hmax = anyh ? __uint_as_float(scal[0]) : FARF;
  const float smin = anyh ? __uint_as_float(scal[1]) : FARF;
  const float seff = hit ? s4 : hmax;
  const float den  = fmaxf(hmax - smin, 1e-7f);
  const float sn   = (seff - smin) / den;
#pragma unroll
  for (int c = 0; c < 3; ++c) {
    const float v = hit ? fminf(fmaxf(ws[i * 4 + c] * sn, 0.0f), 1.0f) : bg[c];
    out[c * N_RAYS + i] = v;
  }
}

__global__ void sdf_init(unsigned* scal) {
  if (threadIdx.x == 0) {
    scal[0] = 0u;           // max neutral
    scal[1] = 0x7f800000u;  // +inf (min neutral)
    scal[2] = 0u;           // any_hit
  }
}

extern "C" void kernel_launch(void* const* d_in, const int* in_sizes, int n_in,
                              void* d_out, int out_size, void* d_ws, size_t ws_size,
                              hipStream_t stream) {
  const float* eye = (const float*)d_in[0];
  const float* cam = (const float*)d_in[1];
  const float* rays = (const float*)d_in[2];
  const float* bg  = (const float*)d_in[3];
  const float* dW1 = (const float*)d_in[4];
  const float* db1 = (const float*)d_in[5];
  const float* dW2 = (const float*)d_in[6];
  const float* db2 = (const float*)d_in[7];
  const float* dW3 = (const float*)d_in[8];
  const float* db3 = (const float*)d_in[9];
  const float* tW1 = (const float*)d_in[10];
  const float* tb1 = (const float*)d_in[11];
  const float* tW2 = (const float*)d_in[12];
  const float* tb2 = (const float*)d_in[13];
  float* outp = (float*)d_out;

  float* wsf = (float*)d_ws;                       // N*4 floats per-ray
  unsigned* scal = (unsigned*)(wsf + 4 * N_RAYS);  // + 3 reduction scalars

  sdf_init<<<1, 64, 0, stream>>>(scal);
  sdf_pass1<<<N_RAYS / 256, 256, 0, stream>>>(
      eye, cam, rays, dW1, db1, dW2, db2, dW3, db3,
      tW1, tb1, tW2, tb2, wsf, scal);
  sdf_pass2<<<(N_RAYS + 255) / 256, 256, 0, stream>>>(wsf, scal, bg, outp);
}
